// Net_35940286333623
// MI455X (gfx1250) — matrix operand order repaired and checked by reading, not compiled, not run on hardware
//
#include <hip/hip_runtime.h>
#include <hip/hip_bf16.h>

// Problem constants (from reference setup_inputs)
#define BB 2
#define DD 96
#define HH 96
#define WW 96
#define LL (HH*WW)        // 9216
#define KK 4
#define RR 6
#define NN 16
#define CC (RR + 2*NN)    // 38
#define CL 128            // scan chunk length
#define NC (LL/CL)        // 72 chunks

typedef __attribute__((ext_vector_type(16))) __bf16 v16bf;
typedef __attribute__((ext_vector_type(8)))  float  v8f;

#define LOG2E 1.4426950408889634f

// ---------------------------------------------------------------------------
// Kernel 1: cross-scan gather  x(B,D,H,W) -> xs(B,K,D,L) in f32 AND bf16.
//   k=0: hw order; k=1: wh order; k=2: hw reversed; k=3: wh reversed
// The bf16 copy feeds the WMMA B operand with per-element loads strided by L.
// ---------------------------------------------------------------------------
__global__ void k_gather(const float* __restrict__ x, float* __restrict__ xs,
                         __bf16* __restrict__ xsb) {
  size_t idx = (size_t)blockIdx.x * blockDim.x + threadIdx.x;
  const size_t total = (size_t)BB*KK*DD*LL;
  if (idx >= total) return;
  int l = (int)(idx % LL);
  int d = (int)((idx / LL) % DD);
  int k = (int)((idx / ((size_t)LL*DD)) % KK);
  int b = (int)( idx / ((size_t)LL*DD*KK));
  int lk  = (k & 2) ? (LL-1-l) : l;
  // wh order: position lk corresponds to (w = lk/H, h = lk%H) -> src = h*W + w
  int src = (k & 1) ? ((lk % HH) * WW + (lk / HH)) : lk;
  float v = x[((size_t)b*DD + d)*LL + src];
  xs[idx]  = v;
  xsb[idx] = (__bf16)v;
}

// ---------------------------------------------------------------------------
// Kernel 1b: pre-pack projection weights into WMMA A-fragment layout (bf16).
// Fragment index = ((k*3 + kc)*3 + mt)*32 + lane, 16 halves each.
// A layout (16-bit, 16x32): lane = M (c row), halves cover K interleaved.
// Rows c >= 38 zero-padded here so the GEMM has no predicated loads.
// ---------------------------------------------------------------------------
__global__ void k_packA(const float* __restrict__ Wp, __bf16* __restrict__ apack) {
  int t = blockIdx.x*blockDim.x + threadIdx.x;
  const int TOT = KK*3*3*32;                 // 1152 fragment-lanes
  if (t >= TOT) return;
  int ln = t % 32;
  int mt = (t/32) % 3;
  int kc = (t/96) % 3;
  int k  =  t/288;
  int g = ln >> 4, mrow = ln & 15;
  int c = mt*16 + mrow;
  __bf16* o = apack + (size_t)t*16;
  #pragma unroll
  for (int j=0;j<16;j++) {
    int v = j>>1, p = j&1;
    int kkx = ((v<4)? 2*v : 16+2*(v-4)) + p + 8*g;   // K within chunk of 32
    float val = (c < CC) ? Wp[((size_t)k*CC + c)*DD + kc*32 + kkx] : 0.f;
    o[j] = (__bf16)val;
  }
}

// ---------------------------------------------------------------------------
// Kernel 2: projection GEMM with WMMA (bf16 inputs, f32 accum)
//   xdbl[b,k,c,l] = sum_d W[k,c,d] * xs[b,k,d,l],  c in [0,38)
// One wave per (b,k, group of 32 columns). M = 3 tiles of 16, K = 3x32,
// N = 2 tiles of 16. A fragments are direct vector loads; B is gathered per element.
// ---------------------------------------------------------------------------
__global__ void k_proj(const __bf16* __restrict__ xsb, const __bf16* __restrict__ apack,
                       float* __restrict__ xdbl) {
  const int NGRP = LL/32;                 // 288 column groups
  int wid = blockIdx.x;
  int lg  = wid % NGRP;
  int k   = (wid / NGRP) % KK;
  int b   =  wid / (NGRP*KK);
  int ln  = threadIdx.x;                  // 0..31
  int l0  = lg * 32;
  int g   = ln >> 4;
  int mrow = ln & 15;

  v8f acc[3][2];
  v8f zero = {0.f,0.f,0.f,0.f,0.f,0.f,0.f,0.f};
  #pragma unroll
  for (int m=0;m<3;m++)
    #pragma unroll
    for (int n=0;n<2;n++) acc[m][n]=zero;

  const __bf16* xsbk = xsb + ((size_t)(b*KK+k))*DD*LL;
  const v16bf*  ap   = (const v16bf*)apack;

  #pragma unroll
  for (int kc=0; kc<3; kc++) {
    v16bf afr[3];
    #pragma unroll
    for (int mt=0; mt<3; mt++)
      afr[mt] = ap[((size_t)((k*3 + kc)*3 + mt))*32 + ln];
    #pragma unroll
    for (int nt=0; nt<2; nt++) {
      v16bf bfr;
      #pragma unroll
      for (int s=0; s<16; s++) {
        int kk = ((s<8) ? s : s+8) + 8*g;
        bfr[s] = xsbk[(size_t)(kc*32 + kk)*LL + l0 + nt*16 + mrow];
      }
      #pragma unroll
      for (int mt=0; mt<3; mt++)
        acc[mt][nt] = __builtin_amdgcn_wmma_f32_16x16x32_bf16(
            false, afr[mt], false, bfr, (short)0, acc[mt][nt], false, false);
    }
  }

  // ---- store D: VGPR i -> M = i + 8*(lane>=16), lane%16 = N
  float* obk = xdbl + ((size_t)(b*KK+k))*CC*LL;
  #pragma unroll
  for (int mt=0; mt<3; mt++) {
    #pragma unroll
    for (int nt=0; nt<2; nt++) {
      int col = l0 + nt*16 + mrow;
      #pragma unroll
      for (int i=0;i<8;i++) {
        int c = mt*16 + i + 8*g;
        if (c < CC) obk[(size_t)c*LL + col] = acc[mt][nt][i];
      }
    }
  }
}

// ---------------------------------------------------------------------------
// Kernel 3: scan pass 1 — per-chunk local recurrence with h0 = 0.
// Outputs per (b,k,d,chunk): sum of deltas (chunk decay = exp2(A2*sumDelta))
// and local end state S[n]. Block = (chunk,k,b), 96 threads (one per d).
// dt rows (6) and B rows (16) of xdbl are shared across d -> staged in LDS.
// ---------------------------------------------------------------------------
__global__ void k_scan1(const float* __restrict__ xs, const float* __restrict__ xdbl,
                        const float* __restrict__ wdtp, const float* __restrict__ dtb,
                        const float* __restrict__ alogs,
                        float* __restrict__ sdelta, float* __restrict__ S) {
  __shared__ float lds[(RR+NN)*CL];
  int chunk = blockIdx.x, k = blockIdx.y, b = blockIdx.z;
  int d = threadIdx.x;                       // 0..95
  int c0 = chunk*CL;
  const float* xbk = xdbl + ((size_t)(b*KK+k))*CC*LL;
  for (int i=d; i<(RR+NN)*CL; i+=DD) {
    int row = i/CL, t = i%CL;
    lds[i] = xbk[(size_t)row*LL + c0 + t];
  }
  __syncthreads();

  float wdt[RR];
  #pragma unroll
  for (int r=0;r<RR;r++) wdt[r] = wdtp[((size_t)(k*DD+d))*RR + r];
  float bias = dtb[k*DD+d];
  float A2[NN];
  #pragma unroll
  for (int n=0;n<NN;n++)
    A2[n] = -__expf(alogs[((size_t)(k*DD+d))*NN+n]) * LOG2E;

  const float* up = xs + ((size_t)(b*KK+k)*DD + d)*LL + c0;
  float h[NN];
  #pragma unroll
  for (int n=0;n<NN;n++) h[n]=0.f;
  float sd = 0.f;

  for (int t=0;t<CL;t++) {
    if ((t & 31) == 0) __builtin_prefetch(up + t + 32, 0, 0);
    float dtr = bias;
    #pragma unroll
    for (int r=0;r<RR;r++) dtr += wdt[r]*lds[r*CL+t];
    float delta = (dtr > 20.f) ? dtr : log1pf(__expf(dtr));   // softplus
    sd += delta;
    float du = delta * up[t];
    #pragma unroll
    for (int n=0;n<NN;n++)
      h[n] = __builtin_amdgcn_exp2f(delta*A2[n])*h[n] + du*lds[(RR+n)*CL+t];
  }
  size_t ci = ((size_t)(b*KK+k)*DD + d)*NC + chunk;
  sdelta[ci] = sd;
  #pragma unroll
  for (int n=0;n<NN;n++) S[ci*NN + n] = h[n];
}

// ---------------------------------------------------------------------------
// Kernel 4: scan pass 2 — sequential combine over chunks (tiny).
// One thread per (b,k,d,n). Hin[chunk] = state entering that chunk.
// ---------------------------------------------------------------------------
__global__ void k_scan2(const float* __restrict__ sdelta, const float* __restrict__ S,
                        const float* __restrict__ alogs, float* __restrict__ Hin) {
  int idx = blockIdx.x*blockDim.x + threadIdx.x;
  if (idx >= BB*KK*DD*NN) return;
  int n = idx % NN;
  int d = (idx/NN) % DD;
  int k = (idx/(NN*DD)) % KK;
  int b =  idx/(NN*DD*KK);
  float a2 = -__expf(alogs[((size_t)(k*DD+d))*NN+n]) * LOG2E;
  size_t base = ((size_t)(b*KK+k)*DD + d)*NC;
  float Hc = 0.f;
  for (int c=0;c<NC;c++) {
    Hin[(base+c)*NN + n] = Hc;
    Hc = __builtin_amdgcn_exp2f(a2*sdelta[base+c])*Hc + S[(base+c)*NN + n];
  }
}

// ---------------------------------------------------------------------------
// Kernel 5: scan pass 3 — full recurrence seeded with Hin, produce y
// (including the D*u skip term). LDS holds dt(6)+B(16)+C(16) rows.
// ---------------------------------------------------------------------------
__global__ void k_scan3(const float* __restrict__ xs, const float* __restrict__ xdbl,
                        const float* __restrict__ wdtp, const float* __restrict__ dtb,
                        const float* __restrict__ alogs, const float* __restrict__ Dsp,
                        const float* __restrict__ Hin, float* __restrict__ ys) {
  __shared__ float lds[CC*CL];
  int chunk = blockIdx.x, k = blockIdx.y, b = blockIdx.z;
  int d = threadIdx.x;
  int c0 = chunk*CL;
  const float* xbk = xdbl + ((size_t)(b*KK+k))*CC*LL;
  for (int i=d; i<CC*CL; i+=DD) {
    int row = i/CL, t = i%CL;
    lds[i] = xbk[(size_t)row*LL + c0 + t];
  }
  __syncthreads();

  float wdt[RR];
  #pragma unroll
  for (int r=0;r<RR;r++) wdt[r] = wdtp[((size_t)(k*DD+d))*RR + r];
  float bias = dtb[k*DD+d];
  float A2[NN];
  #pragma unroll
  for (int n=0;n<NN;n++)
    A2[n] = -__expf(alogs[((size_t)(k*DD+d))*NN+n]) * LOG2E;
  float Dv = Dsp[k*DD+d];

  size_t ci = ((size_t)(b*KK+k)*DD + d)*NC + chunk;
  float h[NN];
  #pragma unroll
  for (int n=0;n<NN;n++) h[n] = Hin[ci*NN + n];

  const float* up = xs + ((size_t)(b*KK+k)*DD + d)*LL + c0;
  float* yp = ys + ((size_t)(b*KK+k)*DD + d)*LL + c0;

  for (int t=0;t<CL;t++) {
    if ((t & 31) == 0) __builtin_prefetch(up + t + 32, 0, 0);
    float dtr = bias;
    #pragma unroll
    for (int r=0;r<RR;r++) dtr += wdt[r]*lds[r*CL+t];
    float delta = (dtr > 20.f) ? dtr : log1pf(__expf(dtr));
    float u = up[t];
    float du = delta * u;
    float y = Dv * u;
    #pragma unroll
    for (int n=0;n<NN;n++)
      h[n] = __builtin_amdgcn_exp2f(delta*A2[n])*h[n] + du*lds[(RR+n)*CL+t];
    #pragma unroll
    for (int n=0;n<NN;n++)
      y += h[n]*lds[(RR+NN+n)*CL+t];
    yp[t] = y;
  }
}

// ---------------------------------------------------------------------------
// Kernel 6: cross-merge + layernorm over D, output (B,H,W,D)
// ---------------------------------------------------------------------------
__global__ void k_merge(const float* __restrict__ ys, const float* __restrict__ nw,
                        const float* __restrict__ nb, float* __restrict__ out) {
  __shared__ float sbuf[DD];
  int pos = blockIdx.x;
  int l = pos % LL; int b = pos / LL;
  int hh = l / WW, ww = l % WW;
  int lT = ww*HH + hh;                 // this pixel's index in the wh scan
  int d = threadIdx.x;
  size_t base = (size_t)b*KK*DD*LL;
  float v = ys[base + ((size_t)(0*DD + d))*LL + l]
          + ys[base + ((size_t)(2*DD + d))*LL + (LL-1-l)]
          + ys[base + ((size_t)(1*DD + d))*LL + lT]
          + ys[base + ((size_t)(3*DD + d))*LL + (LL-1-lT)];
  sbuf[d] = v;
  __syncthreads();
  float s=0.f, s2=0.f;
  for (int i=0;i<DD;i++) { float t = sbuf[i]; s += t; s2 += t*t; }
  float mu  = s * (1.f/DD);
  float var = s2 * (1.f/DD) - mu*mu;
  out[(size_t)pos*DD + d] = (v-mu)*rsqrtf(var + 1e-5f)*nw[d] + nb[d];
}

// ---------------------------------------------------------------------------
extern "C" void kernel_launch(void* const* d_in, const int* in_sizes, int n_in,
                              void* d_out, int out_size, void* d_ws, size_t ws_size,
                              hipStream_t stream) {
  const float* x   = (const float*)d_in[0];   // (B,D,H,W)
  const float* xpw = (const float*)d_in[1];   // (K,38,D)
  const float* dtw = (const float*)d_in[2];   // (K,D,R)
  const float* dtb = (const float*)d_in[3];   // (K,D)
  const float* alg = (const float*)d_in[4];   // (K*D,N)
  const float* Dsp = (const float*)d_in[5];   // (K*D,)
  const float* nw  = (const float*)d_in[6];   // (D,)
  const float* nb  = (const float*)d_in[7];   // (D,)
  float* out = (float*)d_out;

  char* w = (char*)d_ws;
  float*  xs   = (float*)w;   w += (size_t)BB*KK*DD*LL*sizeof(float);    // 28.3 MB
  float*  xdbl = (float*)w;   w += (size_t)BB*KK*CC*LL*sizeof(float);    // 11.2 MB
  float*  ys   = (float*)w;   w += (size_t)BB*KK*DD*LL*sizeof(float);    // 28.3 MB
  float*  sdel = (float*)w;   w += (size_t)BB*KK*DD*NC*sizeof(float);
  float*  S    = (float*)w;   w += (size_t)BB*KK*DD*NC*NN*sizeof(float);
  float*  Hin  = (float*)w;   w += (size_t)BB*KK*DD*NC*NN*sizeof(float);
  __bf16* xsb  = (__bf16*)w;  w += (size_t)BB*KK*DD*LL*sizeof(__bf16);   // 14.2 MB
  __bf16* apack= (__bf16*)w;  w += (size_t)KK*3*3*32*16*sizeof(__bf16);  // 36 KB

  size_t tot = (size_t)BB*KK*DD*LL;
  k_gather<<<(unsigned)((tot+255)/256), 256, 0, stream>>>(x, xs, xsb);
  k_packA<<<(KK*3*3*32 + 255)/256, 256, 0, stream>>>(xpw, apack);
  k_proj<<<BB*KK*(LL/32), 32, 0, stream>>>(xsb, apack, xdbl);
  dim3 g1(NC, KK, BB);
  k_scan1<<<g1, DD, 0, stream>>>(xs, xdbl, dtw, dtb, alg, sdel, S);
  k_scan2<<<(BB*KK*DD*NN + 255)/256, 256, 0, stream>>>(sdel, S, alg, Hin);
  k_scan3<<<g1, DD, 0, stream>>>(xs, xdbl, dtw, dtb, alg, Dsp, Hin, ys);
  k_merge<<<BB*LL, DD, 0, stream>>>(ys, nw, nb, out);
}
